// ODEBlock_82282983457002
// MI455X (gfx1250) — compile-verified, matrix operand order checked by reading, not run on hardware
//
#include <hip/hip_runtime.h>
#include <cmath>
#include <cstdint>

typedef __attribute__((ext_vector_type(16))) _Float16 v16h;
typedef __attribute__((ext_vector_type(8)))  _Float16 v8h;
typedef __attribute__((ext_vector_type(8)))  float    v8f;
typedef __attribute__((ext_vector_type(4)))  int      v4i;

#define N_NODES 50000
#define FEAT    16
#define T_HIST  10
#define T_OUTN  10
#define E_EDGES 800000
#define DE      4
#define HID     64
#define K_SUB   4
#define AUGF    176     // (T_HIST-1)*F + 2F
#define KE      64      // edge MLP K padded (36 valid + bias row 36 -> 64)
#define KN      224     // node MLP K padded (209 valid + bias row 209 -> 224)
#define NF      (N_NODES * FEAT)

#if defined(__gfx1250__) && __has_builtin(__builtin_amdgcn_global_load_async_to_lds_b128)
#define HAVE_ASYNC_LDS 1
#else
#define HAVE_ASYNC_LDS 0
#endif

// Dormand-Prince tableau
__constant__ float cA[5][5] = {
    {0.2f, 0.f, 0.f, 0.f, 0.f},
    {3.f/40.f, 9.f/40.f, 0.f, 0.f, 0.f},
    {44.f/45.f, -56.f/15.f, 32.f/9.f, 0.f, 0.f},
    {19372.f/6561.f, -25360.f/2187.f, 64448.f/6561.f, -212.f/729.f, 0.f},
    {9017.f/3168.f, -355.f/33.f, 46732.f/5247.f, 49.f/176.f, -5103.f/18656.f}};
__constant__ float cB[6] = {35.f/384.f, 0.f, 500.f/1113.f, 125.f/192.f, -2187.f/6784.f, 11.f/84.f};
__constant__ float cC[6] = {0.f, 0.2f, 0.3f, 0.8f, 8.f/9.f, 1.f};

// Hardware tanh on the TRANS pipe (co-executes with WMMA); libm fallback.
__device__ __forceinline__ float fast_tanh(float x) {
#if defined(__gfx1250__) && __has_builtin(__builtin_amdgcn_tanhf)
    return __builtin_amdgcn_tanhf(x);
#elif defined(__gfx1250__) && __has_builtin(__builtin_amdgcn_tanh_f32)
    return __builtin_amdgcn_tanh_f32(x);
#else
    return tanhf(x);
#endif
}

// Async global->LDS 16-byte copy (async data path, ASYNCcnt-tracked).
// Builtin signature (per hipcc diagnostic): non-const AS(1) v4i* src, AS(3) v4i* dst.
__device__ __forceinline__ void cp16_async(_Float16* ldst, const _Float16* gsrc) {
#if HAVE_ASYNC_LDS
    typedef __attribute__((address_space(1))) v4i* gas_p;
    typedef __attribute__((address_space(3))) v4i* las_p;
    __builtin_amdgcn_global_load_async_to_lds_b128(
        (gas_p)(unsigned long long)gsrc,
        (las_p)(unsigned)(unsigned long long)ldst, 0, 0);
#else
    *(uint4*)ldst = *(const uint4*)gsrc;
#endif
}

__device__ __forceinline__ void wait_async_lds() {
#if HAVE_ASYNC_LDS
#if __has_builtin(__builtin_amdgcn_s_wait_asynccnt)
    __builtin_amdgcn_s_wait_asynccnt(0);
#else
    asm volatile("s_wait_asynccnt 0" ::: "memory");
#endif
#endif
}

// Load one 16x32 f16 A/B fragment (ISA 7.12.2 layout): lane<16 -> row=lane,
// halves 0..7 = K[32kb..+7], halves 8..15 = K[32kb+16..+23]; lane>=16 shifts +8.
__device__ __forceinline__ v16h load_frag(const _Float16* tile, int stride_h, int lane, int kb) {
    const _Float16* p = tile + (lane & 15) * stride_h + kb * 32 + ((lane & 16) ? 8 : 0);
    v16h f;
    *(v8h*)&f       = *(const v8h*)(p);
    *((v8h*)&f + 1) = *(const v8h*)(p + 16);
    return f;
}

__device__ __forceinline__ v8f wmma16(v16h a, v16h b, v8f c) {
    return __builtin_amdgcn_wmma_f32_16x16x32_f16(false, a, false, b, (short)0, c, false, false);
}

// ---------------------------------------------------------------- weights ----
// Transposed f16 weights with the layer-1 bias folded in as an extra K row.
__global__ void k_prep_w(const float* __restrict__ w1m, const float* __restrict__ b1m,
                         const float* __restrict__ w2m,
                         const float* __restrict__ w1n, const float* __restrict__ b1n,
                         const float* __restrict__ w2n,
                         _Float16* __restrict__ w1mt, _Float16* __restrict__ w2mt,
                         _Float16* __restrict__ w1nt, _Float16* __restrict__ w2nt) {
    const int S1 = 64 * KE, S2 = 16 * HID, S3 = 64 * KN, S4 = 16 * HID;
    for (int i = blockIdx.x * blockDim.x + threadIdx.x; i < S1 + S2 + S3 + S4;
         i += gridDim.x * blockDim.x) {
        if (i < S1) {                       // W1m^T: [64 hid][64 K]
            int n = i / KE, k = i % KE;
            float v = (k < 36) ? w1m[k * HID + n] : (k == 36 ? b1m[n] : 0.f);
            w1mt[i] = (_Float16)v;
        } else if (i < S1 + S2) {           // W2m^T: [16 F][64 H]
            int j = i - S1; int f = j / HID, h = j % HID;
            w2mt[j] = (_Float16)w2m[h * FEAT + f];
        } else if (i < S1 + S2 + S3) {      // W1n^T: [64 hid][224 K]
            int j = i - S1 - S2; int n = j / KN, k = j % KN;
            float v = (k < 209) ? w1n[k * HID + n] : (k == 209 ? b1n[n] : 0.f);
            w1nt[j] = (_Float16)v;
        } else {                            // W2n^T: [16 F][64 H]
            int j = i - S1 - S2 - S3; int f = j / HID, h = j % HID;
            w2nt[j] = (_Float16)w2n[h * FEAT + f];
        }
    }
}

// ------------------------------------------------------- static node feats ----
__global__ void k_aug(const float* __restrict__ xh, const float* __restrict__ xm,
                      _Float16* __restrict__ augh, float* __restrict__ xcur) {
    int i = blockIdx.x * blockDim.x + threadIdx.x;
    if (i >= NF) return;
    const int n = i / FEAT, f = i % FEAT;
    float xs[T_HIST], ms[T_HIST], cnt = 0.f, sum = 0.f;
#pragma unroll
    for (int t = 0; t < T_HIST; ++t) {
        xs[t] = xh[(size_t)t * NF + n * FEAT + f];
        ms[t] = xm[t * N_NODES + n];
        cnt += ms[t]; sum += xs[t] * ms[t];
    }
#pragma unroll
    for (int t = 0; t < T_HIST - 1; ++t)
        augh[(size_t)n * AUGF + t * FEAT + f] =
            (_Float16)((xs[t + 1] - xs[t]) * (ms[t + 1] * ms[t]));
    float c = cnt < 1.f ? 1.f : cnt;
    float mean = sum / c, v = 0.f;
#pragma unroll
    for (int t = 0; t < T_HIST; ++t) { float d = xs[t] - mean; v += d * d * ms[t]; }
    augh[(size_t)n * AUGF + 144 + f] = (_Float16)mean;
    augh[(size_t)n * AUGF + 160 + f] = (_Float16)(v / c);
    xcur[i] = xs[T_HIST - 1];
}

// ------------------------------------------------------------ RK stage prep ---
__global__ void k_stage_prep(const float* __restrict__ xcur, const float* __restrict__ kbuf,
                             const float* __restrict__ t, int step, int stage,
                             _Float16* __restrict__ xih, float* __restrict__ agg) {
    int i = blockIdx.x * blockDim.x + threadIdx.x;
    if (i >= NF) return;
    float acc = xcur[i];
    if (stage > 0) {
        const int g = step / K_SUB;
        const float dti = (t[g + 1] - t[g]) * (1.f / K_SUB);
        for (int j = 0; j < stage; ++j)
            acc += dti * cA[stage - 1][j] * kbuf[j * NF + i];
    }
    xih[i] = (_Float16)acc;
    agg[i] = 0.f;
}

// ---------------------------------------------------------------- edge MLP ----
// One wave = 16 edges. GEMM1 transposed: h1^T = W1m^T (64x64) @ m^T (64x16).
__global__ void __launch_bounds__(256) k_edge(
        const _Float16* __restrict__ xh, const int* __restrict__ src,
        const int* __restrict__ dst, const float* __restrict__ eattr,
        const _Float16* __restrict__ w1t, const _Float16* __restrict__ w2t,
        const float* __restrict__ b2,
        float* __restrict__ agg) {
    __shared__ __align__(16) _Float16 smem[8 * 16 * KE];
    const int wave = threadIdx.x >> 5, lane = threadIdx.x & 31;
    const int tile = blockIdx.x * 8 + wave;
    const bool active = tile < (E_EDGES / 16);
    _Float16* my = smem + wave * 16 * KE;
    const int ebase = tile * 16;

    if (active) {   // build m^T tile in LDS: [16 edges][64 K] f16
        const int e = ebase + (lane & 15);
        const int idx = (lane < 16) ? src[e] : dst[e];
        const int boff = (lane < 16) ? 0 : 16;        // K half-offset in row
        const _Float16* g = xh + idx * FEAT;
        _Float16* l = my + (lane & 15) * KE + boff;
        cp16_async(l, g);                              // halves 0..7
        cp16_async(l + 8, g + 8);                      // halves 8..15
        if (lane < 16) {                               // attr + bias-one + pad
            const float* ap = eattr + e * DE;
            _Float16* row = my + lane * KE;
            row[32] = (_Float16)ap[0]; row[33] = (_Float16)ap[1];
            row[34] = (_Float16)ap[2]; row[35] = (_Float16)ap[3];
            row[36] = (_Float16)1.f;                   // bias row input
            row[37] = (_Float16)0.f; row[38] = (_Float16)0.f; row[39] = (_Float16)0.f;
            uint4 z = {0u, 0u, 0u, 0u};
            *(uint4*)(row + 40) = z; *(uint4*)(row + 48) = z; *(uint4*)(row + 56) = z;
        }
    }
    wait_async_lds();
    __syncthreads();
    if (!active) return;

    v8f acc[4] = {};
#pragma unroll
    for (int kb = 0; kb < 2; ++kb) {
        v16h b = load_frag(my, KE, lane, kb);
#pragma unroll
        for (int m = 0; m < 4; ++m) {
            v16h a = load_frag(w1t + m * 16 * KE, KE, lane, kb);
            acc[m] = wmma16(a, b, acc[m]);
        }
    }
    const int half8 = (lane & 16) ? 8 : 0;
#pragma unroll
    for (int m = 0; m < 4; ++m)
#pragma unroll
        for (int j = 0; j < 8; ++j)
            acc[m][j] = fast_tanh(acc[m][j]);          // bias already folded in
    // f32 D accumulators already match GEMM2 B-fragment striping: just narrow.
    v16h p0, p1;
#pragma unroll
    for (int j = 0; j < 8; ++j) {
        p0[j] = (_Float16)acc[0][j]; p0[8 + j] = (_Float16)acc[1][j];
        p1[j] = (_Float16)acc[2][j]; p1[8 + j] = (_Float16)acc[3][j];
    }
    v8f out = {};
    out = wmma16(load_frag(w2t, HID, lane, 0), p0, out);
    out = wmma16(load_frag(w2t, HID, lane, 1), p1, out);
    const int d = dst[ebase + (lane & 15)];
    float* ag = agg + d * FEAT + half8;
#pragma unroll
    for (int j = 0; j < 8; ++j) atomicAdd(ag + j, out[j] + b2[half8 + j]);
}

// ---------------------------------------------------------------- node MLP ----
// One wave = 16 nodes. h = [x(16), agg(16), aug(176), t(1), one(1), pad] K=224.
__global__ void __launch_bounds__(128) k_node(
        const _Float16* __restrict__ xh, const float* __restrict__ agg,
        const _Float16* __restrict__ augh,
        const _Float16* __restrict__ w1t, const _Float16* __restrict__ w2t,
        const float* __restrict__ b2,
        const float* __restrict__ t, int step, int stage,
        float* __restrict__ kout) {
    __shared__ __align__(16) _Float16 smem[4 * 16 * KN];
    const int wave = threadIdx.x >> 5, lane = threadIdx.x & 31;
    const int tile = blockIdx.x * 4 + wave;
    const bool active = tile < (N_NODES / 16);
    _Float16* my = smem + wave * 16 * KN;
    const int g = step / K_SUB;
    const float tg = t[g];
    const float dti = (t[g + 1] - tg) * (1.f / K_SUB);
    const _Float16 tih = (_Float16)(tg + (step % K_SUB) * dti + cC[stage] * dti);
    const int nb = tile * 16;

    if (active) {   // build h^T tile in LDS: [16 nodes][224 K], 28 16B chunks/row
        for (int c = lane; c < 16 * 28; c += 32) {
            const int row = c / 28, ch = c % 28, node = nb + row;
            _Float16* dp = my + row * KN + ch * 8;
            if (ch < 2) {
                cp16_async(dp, xh + node * FEAT + ch * 8);
            } else if (ch < 4) {
                const float* ap = agg + node * FEAT + (ch - 2) * 8;
#pragma unroll
                for (int j = 0; j < 8; ++j) dp[j] = (_Float16)ap[j];
            } else if (ch < 26) {
                cp16_async(dp, augh + (size_t)node * AUGF + (ch - 4) * 8);
            } else if (ch == 26) {
                dp[0] = tih;
                dp[1] = (_Float16)1.f;                 // bias row input (K=209)
#pragma unroll
                for (int j = 2; j < 8; ++j) dp[j] = (_Float16)0.f;
            } else {
                uint4 z = {0u, 0u, 0u, 0u}; *(uint4*)dp = z;
            }
        }
    }
    wait_async_lds();
    __syncthreads();
    if (!active) return;

    v8f acc[4] = {};
    for (int kb = 0; kb < 7; ++kb) {
        v16h b = load_frag(my, KN, lane, kb);
#pragma unroll
        for (int m = 0; m < 4; ++m) {
            v16h a = load_frag(w1t + m * 16 * KN, KN, lane, kb);
            acc[m] = wmma16(a, b, acc[m]);
        }
    }
    const int half8 = (lane & 16) ? 8 : 0;
#pragma unroll
    for (int m = 0; m < 4; ++m)
#pragma unroll
        for (int j = 0; j < 8; ++j)
            acc[m][j] = fast_tanh(acc[m][j]);
    v16h p0, p1;
#pragma unroll
    for (int j = 0; j < 8; ++j) {
        p0[j] = (_Float16)acc[0][j]; p0[8 + j] = (_Float16)acc[1][j];
        p1[j] = (_Float16)acc[2][j]; p1[8 + j] = (_Float16)acc[3][j];
    }
    v8f out = {};
    out = wmma16(load_frag(w2t, HID, lane, 0), p0, out);
    out = wmma16(load_frag(w2t, HID, lane, 1), p1, out);
    float* op = kout + (nb + (lane & 15)) * FEAT + half8;
#pragma unroll
    for (int j = 0; j < 8; ++j) op[j] = out[j] + b2[half8 + j];
}

// ------------------------------------------------------------------ combine ---
__global__ void k_combine(float* __restrict__ xcur, const float* __restrict__ kbuf,
                          const float* __restrict__ t, int step, float* __restrict__ ys) {
    int i = blockIdx.x * blockDim.x + threadIdx.x;
    if (i >= NF) return;
    const int g = step / K_SUB;
    const float dti = (t[g + 1] - t[g]) * (1.f / K_SUB);
    float x = xcur[i];
#pragma unroll
    for (int j = 0; j < 6; ++j) x += dti * cB[j] * kbuf[j * NF + i];
    xcur[i] = x;
    if ((step & 3) == 3) ys[(size_t)(step / K_SUB) * NF + i] = x;
}

__global__ void k_gather(const float* __restrict__ ys, const int* __restrict__ midx,
                         float* __restrict__ out) {
    int i = blockIdx.x * blockDim.x + threadIdx.x;
    if (i >= T_OUTN * NF) return;
    int p = i / NF, r = i % NF;
    out[i] = ys[(size_t)midx[p] * NF + r];
}

// -------------------------------------------------------------------- launch --
extern "C" void kernel_launch(void* const* d_in, const int* in_sizes, int n_in,
                              void* d_out, int out_size, void* d_ws, size_t ws_size,
                              hipStream_t stream) {
    const float* x_hist = (const float*)d_in[0];
    const float* x_mask = (const float*)d_in[1];
    const int*   eidx   = (const int*)d_in[2];
    const float* eattr  = (const float*)d_in[3];
    const float* t      = (const float*)d_in[4];
    const int*   midx   = (const int*)d_in[5];
    const float* W1m = (const float*)d_in[6],  *b1m = (const float*)d_in[7];
    const float* W2m = (const float*)d_in[8],  *b2m = (const float*)d_in[9];
    const float* W1n = (const float*)d_in[10], *b1n = (const float*)d_in[11];
    const float* W2n = (const float*)d_in[12], *b2n = (const float*)d_in[13];
    const int* src = eidx;
    const int* dst = eidx + E_EDGES;

    char* p = (char*)d_ws;
    auto carve = [&](size_t bytes) { char* r = p; p += (bytes + 255) & ~(size_t)255; return r; };
    float*     xcur = (float*)carve((size_t)NF * 4);
    float*     agg  = (float*)carve((size_t)NF * 4);
    float*     kbuf = (float*)carve((size_t)6 * NF * 4);
    float*     ys   = (float*)carve((size_t)T_OUTN * NF * 4);
    _Float16*  xih  = (_Float16*)carve((size_t)NF * 2);
    _Float16*  augh = (_Float16*)carve((size_t)N_NODES * AUGF * 2);
    _Float16*  w1mt = (_Float16*)carve((size_t)64 * KE * 2);
    _Float16*  w2mt = (_Float16*)carve((size_t)16 * HID * 2);
    _Float16*  w1nt = (_Float16*)carve((size_t)64 * KN * 2);
    _Float16*  w2nt = (_Float16*)carve((size_t)16 * HID * 2);

    k_prep_w<<<40, 256, 0, stream>>>(W1m, b1m, W2m, W1n, b1n, W2n, w1mt, w2mt, w1nt, w2nt);
    k_aug<<<(NF + 255) / 256, 256, 0, stream>>>(x_hist, x_mask, augh, xcur);

    const int eblocks = (E_EDGES / 16) / 8;            // 6250
    const int nblocks = ((N_NODES / 16) + 3) / 4;      // 782
    for (int step = 0; step < T_OUTN * K_SUB; ++step) {
        for (int st = 0; st < 6; ++st) {
            k_stage_prep<<<(NF + 255) / 256, 256, 0, stream>>>(xcur, kbuf, t, step, st, xih, agg);
            k_edge<<<eblocks, 256, 0, stream>>>(xih, src, dst, eattr, w1mt, w2mt, b2m, agg);
            k_node<<<nblocks, 128, 0, stream>>>(xih, agg, augh, w1nt, w2nt, b2n,
                                                t, step, st, kbuf + st * NF);
        }
        k_combine<<<(NF + 255) / 256, 256, 0, stream>>>(xcur, kbuf, t, step, ys);
    }
    k_gather<<<(T_OUTN * NF + 255) / 256, 256, 0, stream>>>(ys, midx, (float*)d_out);
}
